// LSTMMemoryCell_81930796139238
// MI455X (gfx1250) — matrix operand order repaired and checked by reading, not compiled, not run on hardware
//
#include <hip/hip_runtime.h>

// ---------------------------------------------------------------------------
// LSTM cell, MI455X (gfx1250): f32 -> f16 pack, LDS double-buffered WMMA GEMM
// (v_wmma_f32_16x16x32_f16, f32 accum), fused sigmoid/tanh epilogue.
// Stage fetch uses GLOBAL_LOAD_ASYNC_TO_LDS_B128 (ASYNCcnt) when the
// toolchain exposes the builtin; otherwise register-staged global->LDS.
// ---------------------------------------------------------------------------

typedef _Float16 half4  __attribute__((ext_vector_type(4)));
typedef _Float16 half8  __attribute__((ext_vector_type(8)));
typedef _Float16 half16 __attribute__((ext_vector_type(16)));
typedef float    f32x4  __attribute__((ext_vector_type(4)));
typedef float    f32x8  __attribute__((ext_vector_type(8)));
typedef int      i32x4v __attribute__((vector_size(16)));   // matches builtin param

#define BATCH   8192
#define NIN     2048
#define NH      2048
#define KDIM    4096          // NIN + NH
#define BM      128           // block rows (batch)
#define BN      64            // block cols (hidden units), x4 gates
#define KSTEP   32            // K per LDS stage (one WMMA K)
#define LDSROW  40            // 32 data halves + 8 pad halves (bank-conflict free)
#define AROWS   128           // A rows per stage
#define BROWS   256           // 4 gates * 64 cols
#define STROWS  (AROWS + BROWS)
#define NSTAGES (KDIM / KSTEP)

#if defined(__AMDGCN__) && \
    __has_builtin(__builtin_amdgcn_global_load_async_to_lds_b128) && \
    __has_builtin(__builtin_amdgcn_s_wait_asynccnt)
#define USE_ASYNC_LDS 1
#else
#define USE_ASYNC_LDS 0
#endif

#define AS1 __attribute__((address_space(1)))
#define AS3 __attribute__((address_space(3)))

// Global side: builtin wants `int4 AS1*` (prints as "int __device__ *").
#define G4(p)  ((AS1 i32x4v*)(AS1 void*)(void*)(p))
// LDS side: int4 AS3* converts to either `AS3 int4*` or `AS3 void*` params.
#define L4(p)  ((AS3 i32x4v*)(AS3 void*)(p))

union H16 { half16 v; half8 h[2]; };

__device__ __forceinline__ float sigmoid_fast(float x) {
  return 1.0f / (1.0f + __expf(-x));
}
__device__ __forceinline__ float tanh_fast(float x) {
  // 1 - 2/(e^{2x}+1): correct saturation at +-inf
  float e = __expf(2.0f * x);
  return 1.0f - 2.0f / (e + 1.0f);
}

// ---- pack [x | h_prev] -> f16 combined[8192 x 4096] -----------------------
__global__ __launch_bounds__(256) void pack_xh(const float* __restrict__ x,
                                               const float* __restrict__ h,
                                               _Float16* __restrict__ dst) {
  int idx = (blockIdx.x * 256 + threadIdx.x) * 4;
  int row = idx >> 12;      // / 4096
  int col = idx & 4095;
  const float* s = (col < NIN) ? (x + (size_t)row * NIN + col)
                               : (h + (size_t)row * NH + (col - NIN));
  f32x4 v = *(const f32x4*)s;
  half4 o = { (_Float16)v.x, (_Float16)v.y, (_Float16)v.z, (_Float16)v.w };
  *(half4*)(dst + idx) = o;
}

// ---- pack W_f/i/c/o -> f16 gate-major W[8192 x 4096] ----------------------
__global__ __launch_bounds__(256) void pack_w(const float* __restrict__ wf,
                                              const float* __restrict__ wi,
                                              const float* __restrict__ wc,
                                              const float* __restrict__ wo,
                                              _Float16* __restrict__ dst) {
  int idx = (blockIdx.x * 256 + threadIdx.x) * 4;
  int row = idx >> 12;
  int col = idx & 4095;
  int g = row >> 11;        // gate 0..3 (f,i,c,o)
  int r = row & 2047;
  const float* w = (g == 0) ? wf : (g == 1) ? wi : (g == 2) ? wc : wo;
  f32x4 v = *(const f32x4*)(w + (size_t)r * KDIM + col);
  half4 o = { (_Float16)v.x, (_Float16)v.y, (_Float16)v.z, (_Float16)v.w };
  *(half4*)(dst + idx) = o;
}

// ---- fused GEMM (4 gates) + LSTM epilogue ---------------------------------
__global__ __launch_bounds__(256) void lstm_gemm_wmma(
    const _Float16* __restrict__ comb,   // [8192, 4096] f16
    const _Float16* __restrict__ wmat,   // [8192, 4096] f16, gate-major rows
    const float* __restrict__ c_prev,
    const float* __restrict__ b_f, const float* __restrict__ b_i,
    const float* __restrict__ b_c, const float* __restrict__ b_o,
    float* __restrict__ h_out, float* __restrict__ c_out) {

  __shared__ _Float16 smem[2][STROWS * LDSROW];   // 60 KB

  const int tid   = threadIdx.x;
  const int mrow0 = blockIdx.y * BM;
  const int ncol0 = blockIdx.x * BN;

  // Stage loader: 1536 16B chunks per stage, 6 per thread (coalesced).
  const _Float16* gptr[6];
  int loff[6];
#pragma unroll
  for (int j = 0; j < 6; ++j) {
    int c    = j * 256 + tid;
    int rowc = c >> 2;              // 4 chunks (64B) per 32-half row
    int q    = c & 3;
    loff[j]  = rowc * LDSROW + q * 8;
    if (rowc < AROWS) {
      gptr[j] = comb + (size_t)(mrow0 + rowc) * KDIM + q * 8;
    } else {
      int r2 = rowc - AROWS;        // 0..255
      int g  = r2 >> 6;             // gate
      int nc = r2 & 63;             // col within tile
      gptr[j] = wmat + (size_t)(g * NH + ncol0 + nc) * KDIM + q * 8;
    }
  }

  const int lane = tid & 31;
  const int l16  = lane & 15;
  const int hi   = lane >> 4;       // half-wave select
  const int wave = tid >> 5;
  const int wm   = wave & 3;        // 4 waves along M -> 32 rows each
  const int wn2  = wave >> 2;       // 2 waves along N -> 32 cols each

  f32x8 acc[2][2][4];
  const f32x8 zero = {0.f, 0.f, 0.f, 0.f, 0.f, 0.f, 0.f, 0.f};
#pragma unroll
  for (int mi = 0; mi < 2; ++mi)
#pragma unroll
    for (int ni = 0; ni < 2; ++ni)
#pragma unroll
      for (int g = 0; g < 4; ++g) acc[mi][ni][g] = zero;

  // ---- prologue: stage 0 -> smem[0] ----
#if USE_ASYNC_LDS
#pragma unroll
  for (int j = 0; j < 6; ++j) {
    __builtin_amdgcn_global_load_async_to_lds_b128(
        G4(gptr[j]), L4(&smem[0][loff[j]]), 0, 0);
  }
  __builtin_amdgcn_s_wait_asynccnt(0);
#else
  {
    half8 r[6];
#pragma unroll
    for (int j = 0; j < 6; ++j) r[j] = *(const half8*)gptr[j];
#pragma unroll
    for (int j = 0; j < 6; ++j) *(half8*)&smem[0][loff[j]] = r[j];
  }
#endif
  __syncthreads();

  for (int ks = 0; ks < NSTAGES; ++ks) {
    const int buf   = ks & 1;
    const bool more = (ks + 1 < NSTAGES);

#if USE_ASYNC_LDS
    // Kick next stage into the other LDS buffer; overlaps all 16 WMMAs below.
    if (more) {
#pragma unroll
      for (int j = 0; j < 6; ++j) {
        __builtin_amdgcn_global_load_async_to_lds_b128(
            G4(gptr[j] + (size_t)(ks + 1) * KSTEP),
            L4(&smem[buf ^ 1][loff[j]]), 0, 0);
      }
    }
#else
    half8 r[6];
    if (more) {
#pragma unroll
      for (int j = 0; j < 6; ++j)
        r[j] = *(const half8*)(gptr[j] + (size_t)(ks + 1) * KSTEP);
    }
#endif
    if (ks + 3 < NSTAGES) {
#pragma unroll
      for (int j = 0; j < 6; ++j)
        __builtin_prefetch(gptr[j] + (size_t)(ks + 3) * KSTEP, 0, 0);
    }

    // A fragments: 16x32 f16. lane L<16: K {0..7,16..23}; L>=16: {8..15,24..31}
    H16 afrag[2];
#pragma unroll
    for (int mi = 0; mi < 2; ++mi) {
      int row = wm * 32 + mi * 16 + l16;
      const _Float16* p = &smem[buf][row * LDSROW];
      afrag[mi].h[0] = *(const half8*)(p + hi * 8);
      afrag[mi].h[1] = *(const half8*)(p + 16 + hi * 8);
    }

    // B fragments: 32x16 f16 per gate/ncol. lane L<16: K {0..7,16..23}; L>=16: {8..15,24..31}
#pragma unroll
    for (int g = 0; g < 4; ++g) {
#pragma unroll
      for (int ni = 0; ni < 2; ++ni) {
        int row = AROWS + g * 64 + wn2 * 32 + ni * 16 + l16;
        const _Float16* p = &smem[buf][row * LDSROW];
        H16 bfrag;
        bfrag.h[0] = *(const half8*)(p + hi * 8);
        bfrag.h[1] = *(const half8*)(p + 16 + hi * 8);
#pragma unroll
        for (int mi = 0; mi < 2; ++mi) {
          acc[mi][ni][g] = __builtin_amdgcn_wmma_f32_16x16x32_f16(
              false, afrag[mi].v, false, bfrag.v,
              (short)0, acc[mi][ni][g], false, false);
        }
      }
    }

#if USE_ASYNC_LDS
    if (more) __builtin_amdgcn_s_wait_asynccnt(0);
#else
    if (more) {
#pragma unroll
      for (int j = 0; j < 6; ++j) *(half8*)&smem[buf ^ 1][loff[j]] = r[j];
    }
#endif
    __syncthreads();
  }

  // ---- fused LSTM epilogue: acc layout VGPR e <-> M = e + 8*hi ------------
#pragma unroll
  for (int ni = 0; ni < 2; ++ni) {
    int n = ncol0 + wn2 * 32 + ni * 16 + l16;
    float bf = b_f[n], bi = b_i[n], bc = b_c[n], bo = b_o[n];
#pragma unroll
    for (int mi = 0; mi < 2; ++mi) {
      int mbase = mrow0 + wm * 32 + mi * 16 + hi * 8;
#pragma unroll
      for (int e = 0; e < 8; ++e) {
        int m = mbase + e;
        size_t off = (size_t)m * NH + n;
        float fl = acc[mi][ni][0][e] + bf;
        float il = acc[mi][ni][1][e] + bi;
        float cl = acc[mi][ni][2][e] + bc;
        float ol = acc[mi][ni][3][e] + bo;
        float ft  = sigmoid_fast(fl);
        float it  = sigmoid_fast(il);
        float ctl = tanh_fast(cl);
        float ot  = sigmoid_fast(ol);
        float cp  = c_prev[off];
        float ct  = ft * cp + it * ctl;
        float ht  = ot * tanh_fast(ct);
        h_out[off] = ht;
        c_out[off] = ct;
      }
    }
  }
}

extern "C" void kernel_launch(void* const* d_in, const int* in_sizes, int n_in,
                              void* d_out, int out_size, void* d_ws, size_t ws_size,
                              hipStream_t stream) {
  (void)in_sizes; (void)n_in; (void)out_size; (void)ws_size;

  const float* x  = (const float*)d_in[0];
  const float* hp = (const float*)d_in[1];
  const float* cp = (const float*)d_in[2];
  const float* Wf = (const float*)d_in[3];
  const float* bf = (const float*)d_in[4];
  const float* Wi = (const float*)d_in[5];
  const float* bi = (const float*)d_in[6];
  const float* Wc = (const float*)d_in[7];
  const float* bc = (const float*)d_in[8];
  const float* Wo = (const float*)d_in[9];
  const float* bo = (const float*)d_in[10];

  float* h_out = (float*)d_out;
  float* c_out = h_out + (size_t)BATCH * NH;

  _Float16* comb = (_Float16*)d_ws;                     // 64 MB
  _Float16* wmat = comb + (size_t)BATCH * KDIM;         // 64 MB

  const int convBlocks = (BATCH * KDIM / 4) / 256;      // 32768
  hipLaunchKernelGGL(pack_xh, dim3(convBlocks), dim3(256), 0, stream, x, hp, comb);
  hipLaunchKernelGGL(pack_w,  dim3(convBlocks), dim3(256), 0, stream,
                     Wf, Wi, Wc, Wo, wmat);

  dim3 grid(NH / BN, BATCH / BM);                       // (32, 64)
  hipLaunchKernelGGL(lstm_gemm_wmma, grid, dim3(256), 0, stream,
                     comb, wmat, cp, bf, bi, bc, bo, h_out, c_out);
}
